// FrankenMACE_72481868087881
// MI455X (gfx1250) — matrix operand order repaired and checked by reading, not compiled, not run on hardware
//
#include <hip/hip_runtime.h>
#include <hip/hip_bf16.h>
#include <hip/hip_fp16.h>

typedef __attribute__((ext_vector_type(16))) _Float16 v16h;
typedef __attribute__((ext_vector_type(8)))  float    v8f;

#define NNODES 16384
#define NEDGES 196608
#define CCH    32
#define SHD    16
#define NBES   8
#define HRAD   64

// fragment-packed weight buffer layout (per layer, in halfs)
constexpr int OFF_W1  = 0;      // 4 frags  (4 N-tiles, K=32 padded from 8)
constexpr int OFF_W2  = 2048;   // 8 frags  (4 N-tiles x 2 K-chunks)
constexpr int OFF_W3  = 6144;   // 8 frags
constexpr int OFF_W4  = 10240;  // 64 frags (32 N-tiles x 2 K-chunks)
constexpr int WFRAG_L = 43008;  // halfs per layer

union AFrag {
  v16h v;
  uint4 q[2];
  _Float16 h[16];
};

// fast silu: v_exp_f32 + v_rcp_f32 + v_mul (no IEEE div chain)
__device__ __forceinline__ float silu(float x) {
  float e = __expf(-x);
  return x * __builtin_amdgcn_rcpf(1.0f + e);
}

// ---------------------------------------------------------------------------
// Kernel 0: node embedding  node_feats[n][c][0] = node_attrs[n] @ W_embed
// ---------------------------------------------------------------------------
__global__ __launch_bounds__(256) void node_init_kernel(
    const float* __restrict__ attrs, const float* __restrict__ Wemb,
    float* __restrict__ nf)
{
  int t = blockIdx.x * 256 + threadIdx.x;     // N*32 threads exactly
  int n = t >> 5, c = t & 31;
  float s = 0.f;
#pragma unroll
  for (int z = 0; z < 8; ++z) s += attrs[n * 8 + z] * Wemb[z * 32 + c];
  float* p = nf + (size_t)n * 512 + c * 16;
  p[0] = s;
#pragma unroll
  for (int i = 1; i < 16; ++i) p[i] = 0.f;
}

// ---------------------------------------------------------------------------
// Kernel 1: edge geometry -> real spherical harmonics (f32) + bessel (f16)
// ---------------------------------------------------------------------------
__global__ __launch_bounds__(256) void geom_kernel(
    const float* __restrict__ pos, const float* __restrict__ shifts,
    const int* __restrict__ ei, float* __restrict__ esph,
    _Float16* __restrict__ erad)
{
  int e = blockIdx.x * 256 + threadIdx.x;
  if (e >= NEDGES) return;
  int sn = ei[e], rc = ei[NEDGES + e];
  float vx = pos[rc * 3 + 0] - pos[sn * 3 + 0] + shifts[e * 3 + 0];
  float vy = pos[rc * 3 + 1] - pos[sn * 3 + 1] + shifts[e * 3 + 1];
  float vz = pos[rc * 3 + 2] - pos[sn * 3 + 2] + shifts[e * 3 + 2];
  float len = sqrtf(vx * vx + vy * vy + vz * vz + 1e-18f);
  float il = __builtin_amdgcn_rcpf(len);
  float x = vx * il, y = vy * il, z = vz * il;
  float x2 = x * x, y2 = y * y, z2 = z * z;

  const float s3 = 1.7320508075688772f;
  const float s15 = 3.8729833462074170f;
  const float s5h = 1.1180339887498949f;
  const float s15h = 1.9364916731037085f;
  const float s35_8 = 2.0916500663351889f;
  const float s105 = 10.246950765959598f;
  const float s21_8 = 1.6201851746019651f;
  const float s7h = 1.3228756555322954f;
  const float s105h = 5.1234753829797990f;

  float sh[16];
  sh[0] = 1.0f;
  sh[1] = s3 * x;  sh[2] = s3 * y;  sh[3] = s3 * z;
  sh[4] = s15 * x * y;  sh[5] = s15 * y * z;
  sh[6] = s5h * (3.f * z2 - 1.f);
  sh[7] = s15 * x * z;
  sh[8] = s15h * (x2 - y2);
  sh[9]  = s35_8 * y * (3.f * x2 - y2);
  sh[10] = s105 * x * y * z;
  sh[11] = s21_8 * y * (5.f * z2 - 1.f);
  sh[12] = s7h * z * (5.f * z2 - 3.f);
  sh[13] = s21_8 * x * (5.f * z2 - 1.f);
  sh[14] = s105h * z * (x2 - y2);
  sh[15] = s35_8 * x * (x2 - 3.f * y2);
#pragma unroll
  for (int j = 0; j < 16; ++j) esph[(size_t)e * 16 + j] = sh[j];

  float u = len * 0.2f;                       // r / R_MAX
  float isr = __builtin_amdgcn_rcpf(fmaxf(len, 1e-9f));
  float u5 = u * u * u * u * u;
  float fc = (u < 1.f) ? (1.f - 21.f * u5 + 35.f * u5 * u - 15.f * u5 * u * u) : 0.f;
  const float pref = 0.63245553203367588f;    // sqrt(2/R_MAX)
  const float pi = 3.14159265358979323846f;
  float g = pref * isr * fc;
#pragma unroll
  for (int n = 1; n <= 8; ++n) {
    float rb = g * __sinf(pi * (float)n * u);
    erad[(size_t)e * 8 + (n - 1)] = (_Float16)rb;
  }
}

// ---------------------------------------------------------------------------
// Kernel 2: pack W1..W4 (both layers) into B-fragment-native f16 layout.
// B-fragment (K x 16): lane holds col N = lane&15; W1: element j -> K = (lane>=16?16:0)+j; W2..W4: K = (lane>=16?8:0)+j+(j&8)
// ---------------------------------------------------------------------------
__global__ __launch_bounds__(256) void prep_weights_kernel(
    const float* __restrict__ W1, const float* __restrict__ W2,
    const float* __restrict__ W3, const float* __restrict__ W4,
    _Float16* __restrict__ wfrag)
{
  int idx = blockIdx.x * 256 + threadIdx.x;   // 2*WFRAG_L threads exactly
  int layer = idx / WFRAG_L;
  int r = idx % WFRAG_L;
  float val = 0.f;
  if (r < OFF_W2) {                                         // W1: 8x64, K-padded
    int frag = r >> 9, rem = r & 511, lane = rem >> 4, j = rem & 15;
    int K = ((lane >> 4) << 4) + j, N = frag * 16 + (lane & 15);
    if (K < NBES) val = W1[layer * NBES * HRAD + K * HRAD + N];
  } else if (r < OFF_W3) {                                  // W2: 64x64
    int rr = r - OFF_W2;
    int frag = rr >> 9, rem = rr & 511, lane = rem >> 4, j = rem & 15;
    int tile = frag >> 1, kc = frag & 1;
    int K = kc * 32 + ((lane >> 4) << 3) + j + (j & 8), N = tile * 16 + (lane & 15);
    val = W2[layer * HRAD * HRAD + K * HRAD + N];
  } else if (r < OFF_W4) {                                  // W3: 64x64
    int rr = r - OFF_W3;
    int frag = rr >> 9, rem = rr & 511, lane = rem >> 4, j = rem & 15;
    int tile = frag >> 1, kc = frag & 1;
    int K = kc * 32 + ((lane >> 4) << 3) + j + (j & 8), N = tile * 16 + (lane & 15);
    val = W3[layer * HRAD * HRAD + K * HRAD + N];
  } else {                                                  // W4: 64x512
    int rr = r - OFF_W4;
    int frag = rr >> 9, rem = rr & 511, lane = rem >> 4, j = rem & 15;
    int tile = frag >> 1, kc = frag & 1;
    int K = kc * 32 + ((lane >> 4) << 3) + j + (j & 8), N = tile * 16 + (lane & 15);
    val = W4[layer * HRAD * 512 + K * 512 + N];
  }
  wfrag[idx] = (_Float16)val;
}

// ---------------------------------------------------------------------------
// Kernel 3: fused edge MLP (WMMA) + message + atomic scatter.
// One wave = 16 edges (one GEMM M-tile). 8 waves/block -> 128 edges/block.
// ---------------------------------------------------------------------------
__global__ __launch_bounds__(256) void edge_mlp_kernel(
    const _Float16* __restrict__ erad, const float* __restrict__ esph,
    const float* __restrict__ nf, const int* __restrict__ ei,
    const _Float16* __restrict__ wfragL, float* __restrict__ agg)
{
  __shared__ __align__(16) _Float16 Hlds[8][16 * HRAD];   // 16 KB
  __shared__ float Slds[8][16][CCH];                      // 16 KB sender scalars
  __shared__ __align__(16) float Sph[8][16][SHD];         //  8 KB
  __shared__ int RcvS[8][16];

  const int wave = threadIdx.x >> 5;
  const int lane = threadIdx.x & 31;
  const int e0 = blockIdx.x * 128 + wave * 16;
  const int row = lane & 15;   // GEMM row (edge) for A, col (N) for B/D
  const int half = lane >> 4;
  const int* snd = ei;
  const int* rcv = ei + NEDGES;

  // gather s[e][c] = node_feats[snd[e]][c][0]; lane = channel
  for (int r = 0; r < 16; ++r) {
    int sn = snd[e0 + r];
    Slds[wave][r][lane] = nf[(size_t)sn * 512 + lane * 16];
  }
  if (lane < 16) {
    int e = e0 + lane;
    RcvS[wave][lane] = rcv[e];
    const float4* sp = reinterpret_cast<const float4*>(esph + (size_t)e * 16);
    float4* wp = reinterpret_cast<float4*>(&Sph[wave][lane][0]);
    wp[0] = sp[0]; wp[1] = sp[1]; wp[2] = sp[2]; wp[3] = sp[3];
  }

  // A-fragment for GEMM1: radial features (K real = 8, rest zero)
  AFrag a1;
#pragma unroll
  for (int j = 0; j < 16; ++j) a1.h[j] = (_Float16)0.f;
  if (lane < 16) {
    const uint4* rp = reinterpret_cast<const uint4*>(erad + (size_t)(e0 + lane) * 8);
    a1.q[0] = rp[0];
  }
  __syncthreads();

  // ---- GEMM1: X(16x32) @ W1 -> H(16x64), silu ----
#pragma unroll
  for (int t = 0; t < 4; ++t) {
    AFrag b;
    const uint4* bp = reinterpret_cast<const uint4*>(wfragL + OFF_W1 + t * 512 + lane * 16);
    b.q[0] = bp[0]; b.q[1] = bp[1];
    v8f acc = {};
    acc = __builtin_amdgcn_wmma_f32_16x16x32_f16(false, a1.v, false, b.v, (short)0, acc, false, false);
#pragma unroll
    for (int r = 0; r < 8; ++r)
      Hlds[wave][(r + half * 8) * HRAD + t * 16 + row] = (_Float16)silu(acc[r]);
  }
  __syncthreads();

  // ---- GEMM2 / GEMM3: H(16x64) @ W(64x64) -> H, silu ----
#pragma unroll
  for (int stage = 0; stage < 2; ++stage) {
    const int offW = (stage == 0) ? OFF_W2 : OFF_W3;
    AFrag a0, a1f;  // K-chunks 0..31 and 32..63
    a0.q[0]  = *reinterpret_cast<const uint4*>(&Hlds[wave][row * HRAD + half * 8]);
    a0.q[1]  = *reinterpret_cast<const uint4*>(&Hlds[wave][row * HRAD + half * 8 + 16]);
    a1f.q[0] = *reinterpret_cast<const uint4*>(&Hlds[wave][row * HRAD + 32 + half * 8]);
    a1f.q[1] = *reinterpret_cast<const uint4*>(&Hlds[wave][row * HRAD + 32 + half * 8 + 16]);
    v8f acc[4];
#pragma unroll
    for (int t = 0; t < 4; ++t) {
      AFrag b0, b1;
      const uint4* bp0 = reinterpret_cast<const uint4*>(wfragL + offW + (t * 2 + 0) * 512 + lane * 16);
      b0.q[0] = bp0[0]; b0.q[1] = bp0[1];
      const uint4* bp1 = reinterpret_cast<const uint4*>(wfragL + offW + (t * 2 + 1) * 512 + lane * 16);
      b1.q[0] = bp1[0]; b1.q[1] = bp1[1];
      v8f zero = {};
      acc[t] = zero;
      acc[t] = __builtin_amdgcn_wmma_f32_16x16x32_f16(false, a0.v, false, b0.v, (short)0, acc[t], false, false);
      acc[t] = __builtin_amdgcn_wmma_f32_16x16x32_f16(false, a1f.v, false, b1.v, (short)0, acc[t], false, false);
    }
    __syncthreads();
#pragma unroll
    for (int t = 0; t < 4; ++t)
#pragma unroll
      for (int r = 0; r < 8; ++r)
        Hlds[wave][(r + half * 8) * HRAD + t * 16 + row] = (_Float16)silu(acc[t][r]);
    __syncthreads();
  }

  // ---- GEMM4: H(16x64) @ W4(64x512) fused with message + atomic scatter ----
  AFrag a40, a41;
  a40.q[0] = *reinterpret_cast<const uint4*>(&Hlds[wave][row * HRAD + half * 8]);
  a40.q[1] = *reinterpret_cast<const uint4*>(&Hlds[wave][row * HRAD + half * 8 + 16]);
  a41.q[0] = *reinterpret_cast<const uint4*>(&Hlds[wave][row * HRAD + 32 + half * 8]);
  a41.q[1] = *reinterpret_cast<const uint4*>(&Hlds[wave][row * HRAD + 32 + half * 8 + 16]);

  int rcvr[8];
  float sphr[8];
#pragma unroll
  for (int r = 0; r < 8; ++r) {
    int erow = r + half * 8;              // edge row this VGPR slot covers
    rcvr[r] = RcvS[wave][erow];
    sphr[r] = Sph[wave][erow][row];       // sph[edge][m], m = D-column = row var
  }

  for (int nt = 0; nt < 32; ++nt) {       // N-tile == output channel c
    AFrag b0, b1;
    const uint4* bp0 = reinterpret_cast<const uint4*>(wfragL + OFF_W4 + (nt * 2 + 0) * 512 + lane * 16);
    b0.q[0] = bp0[0]; b0.q[1] = bp0[1];
    const uint4* bp1 = reinterpret_cast<const uint4*>(wfragL + OFF_W4 + (nt * 2 + 1) * 512 + lane * 16);
    b1.q[0] = bp1[0]; b1.q[1] = bp1[1];
    if (nt + 1 < 32)
      __builtin_prefetch(wfragL + OFF_W4 + (nt + 1) * 2 * 512 + lane * 16, 0, 0);
    v8f acc = {};
    acc = __builtin_amdgcn_wmma_f32_16x16x32_f16(false, a40.v, false, b0.v, (short)0, acc, false, false);
    acc = __builtin_amdgcn_wmma_f32_16x16x32_f16(false, a41.v, false, b1.v, (short)0, acc, false, false);
#pragma unroll
    for (int r = 0; r < 8; ++r) {
      int erow = r + half * 8;
      float sval = Slds[wave][erow][nt];            // s[edge][c]
      float v = acc[r] * sphr[r] * sval;            // tpw * sph * s
      atomicAdd(agg + (size_t)rcvr[r] * 512 + nt * 16 + row, v);
    }
  }
}

// ---------------------------------------------------------------------------
// Kernel 4: node update. 32-thread group per node, thread = output channel o.
// Weight loads hoisted: Wskip is i-invariant, Wlin/Wprod collapse to 4 values
// per c (one per l-slice); the lmap select resolves at compile time.
// ---------------------------------------------------------------------------
__device__ __forceinline__ int lmap(int i) {
  return (i == 0) ? 0 : ((i < 4) ? 1 : ((i < 9) ? 2 : 3));
}

__global__ __launch_bounds__(256) void node_update_kernel(
    const float* __restrict__ agg, float* __restrict__ nf,
    const float* __restrict__ attrs,
    const float* __restrict__ Wlin, const float* __restrict__ Wskip,
    const float* __restrict__ w2v, const float* __restrict__ w3v,
    const float* __restrict__ Wprod, float* __restrict__ outdesc, int layer)
{
  __shared__ __align__(16) float Ag[8][CCH][SHD];
  __shared__ __align__(16) float Nf[8][CCH][SHD];
  __shared__ __align__(16) float Bl[8][CCH][SHD];
  int grp = threadIdx.x >> 5, o = threadIdx.x & 31;
  int node = blockIdx.x * 8 + grp;

#pragma unroll
  for (int i = 0; i < 16; ++i) {
    Ag[grp][o][i] = agg[(size_t)node * 512 + o * 16 + i] * (1.0f / 12.0f);
    Nf[grp][o][i] = nf[(size_t)node * 512 + o * 16 + i];
  }
  __syncthreads();

  int z = 0;
#pragma unroll
  for (int q = 0; q < 8; ++q)
    if (attrs[node * 8 + q] > 0.5f) z = q;
  const float* Ws = Wskip + z * 1024;

  float a[16], sc[16];
#pragma unroll
  for (int i = 0; i < 16; ++i) { a[i] = 0.f; sc[i] = 0.f; }

  for (int c = 0; c < 32; ++c) {
    float wl0 = Wlin[0 * 1024 + c * 32 + o];
    float wl1 = Wlin[1 * 1024 + c * 32 + o];
    float wl2 = Wlin[2 * 1024 + c * 32 + o];
    float wl3 = Wlin[3 * 1024 + c * 32 + o];
    float wsv = Ws[c * 32 + o];
#pragma unroll
    for (int i = 0; i < 16; ++i) {
      float wl = (lmap(i) == 0) ? wl0 : (lmap(i) == 1) ? wl1 : (lmap(i) == 2) ? wl2 : wl3;
      a[i]  += wl  * Ag[grp][c][i];
      sc[i] += wsv * Nf[grp][c][i];
    }
  }

  float inv = a[0], p2 = 0.f;
#pragma unroll
  for (int i = 0; i < 16; ++i) p2 += a[i] * a[i];
  a[0] = inv + w2v[o] * p2 + w3v[o] * inv * p2;
#pragma unroll
  for (int i = 0; i < 16; ++i) Bl[grp][o][i] = a[i];
  __syncthreads();

  float outv[16];
#pragma unroll
  for (int i = 0; i < 16; ++i) outv[i] = sc[i];
  for (int c = 0; c < 32; ++c) {
    float wp0 = Wprod[0 * 1024 + c * 32 + o];
    float wp1 = Wprod[1 * 1024 + c * 32 + o];
    float wp2 = Wprod[2 * 1024 + c * 32 + o];
    float wp3 = Wprod[3 * 1024 + c * 32 + o];
#pragma unroll
    for (int i = 0; i < 16; ++i) {
      float wp = (lmap(i) == 0) ? wp0 : (lmap(i) == 1) ? wp1 : (lmap(i) == 2) ? wp2 : wp3;
      outv[i] += wp * Bl[grp][c][i];
    }
  }
#pragma unroll
  for (int i = 0; i < 16; ++i)
    nf[(size_t)node * 512 + o * 16 + i] = outv[i];
  outdesc[(size_t)node * 64 + layer * 32 + o] = outv[0];
}

// ---------------------------------------------------------------------------
extern "C" void kernel_launch(void* const* d_in, const int* in_sizes, int n_in,
                              void* d_out, int out_size, void* d_ws, size_t ws_size,
                              hipStream_t stream)
{
  (void)in_sizes; (void)n_in; (void)out_size; (void)ws_size;
  const float* atom_pos   = (const float*)d_in[0];
  const float* node_attrs = (const float*)d_in[1];
  const float* shifts     = (const float*)d_in[2];
  const int*   ei         = (const int*)d_in[3];
  const float* W_embed    = (const float*)d_in[4];
  const float* W1         = (const float*)d_in[5];
  const float* W2         = (const float*)d_in[6];
  const float* W3         = (const float*)d_in[7];
  const float* W4         = (const float*)d_in[8];
  const float* Wlin       = (const float*)d_in[9];
  const float* Wskip      = (const float*)d_in[10];
  const float* w2         = (const float*)d_in[11];
  const float* w3         = (const float*)d_in[12];
  const float* Wprod      = (const float*)d_in[13];
  float* out = (float*)d_out;

  char* ws = (char*)d_ws;
  float* nf = (float*)ws;          ws += (size_t)NNODES * 512 * 4;
  float* agg = (float*)ws;         ws += (size_t)NNODES * 512 * 4;
  float* esph = (float*)ws;        ws += (size_t)NEDGES * 16 * 4;
  _Float16* erad = (_Float16*)ws;  ws += (size_t)NEDGES * 8 * 2;
  _Float16* wfrag = (_Float16*)ws; ws += (size_t)2 * WFRAG_L * 2;

  node_init_kernel<<<(NNODES * 32) / 256, 256, 0, stream>>>(node_attrs, W_embed, nf);
  geom_kernel<<<(NEDGES + 255) / 256, 256, 0, stream>>>(atom_pos, shifts, ei, esph, erad);
  prep_weights_kernel<<<(2 * WFRAG_L) / 256, 256, 0, stream>>>(W1, W2, W3, W4, wfrag);

  for (int l = 0; l < 2; ++l) {
    hipMemsetAsync(agg, 0, (size_t)NNODES * 512 * 4, stream);
    edge_mlp_kernel<<<NEDGES / 128, 256, 0, stream>>>(
        erad, esph, nf, ei, wfrag + (size_t)l * WFRAG_L, agg);
    node_update_kernel<<<NNODES / 8, 256, 0, stream>>>(
        agg, nf, node_attrs,
        Wlin + l * 4096, Wskip + l * 8192, w2 + l * 32, w3 + l * 32,
        Wprod + l * 4096, out, l);
  }
}
